// FactorizedAttention_17746804867218
// MI455X (gfx1250) — compile-verified, matrix operand order checked by reading, not run on hardware
//
#include <hip/hip_runtime.h>
#include <hip/hip_bf16.h>

typedef __attribute__((ext_vector_type(16))) __bf16 v16bf;
typedef __attribute__((ext_vector_type(8)))  __bf16 v8bf;
typedef __attribute__((ext_vector_type(8)))  float  v8f;

static __device__ __forceinline__ v8f vzero8f() {
  v8f z;
#pragma unroll
  for (int i = 0; i < 8; ++i) z[i] = 0.0f;
  return z;
}
static __device__ __forceinline__ v16bf ld_v16(const __bf16* p) {
  v8bf lo = *(const v8bf*)(p);
  v8bf hi = *(const v8bf*)(p + 8);
  return __builtin_shufflevector(lo, hi, 0,1,2,3,4,5,6,7,8,9,10,11,12,13,14,15);
}
static __device__ __forceinline__ v8f wmma_bf16(v16bf a, v16bf b, v8f c) {
  return __builtin_amdgcn_wmma_f32_16x16x32_bf16(false, a, false, b, (short)0, c, false, false);
}
// CDNA5 async DMA: per-lane 16B global -> LDS, tracked by ASYNCcnt.
static __device__ __forceinline__ void async_copy_b128(unsigned lds_addr, const void* gaddr) {
  asm volatile("global_load_async_to_lds_b128 %0, %1, off"
               :: "v"(lds_addr), "v"(gaddr) : "memory");
}
static __device__ __forceinline__ void wait_async0() {
  asm volatile("s_wait_asynccnt 0" ::: "memory");
}

// ---------------- tiny zero-fill for conv padding scratch ----------------
__global__ void fa_zero_kernel(float4* p) {
  float4 z; z.x = 0.f; z.y = 0.f; z.z = 0.f; z.w = 0.f;
  p[threadIdx.x] = z;
}

// ---------------- f32 -> bf16 cast ----------------
__global__ void fa_cast_kernel(const float* __restrict__ src, __bf16* __restrict__ dst,
                               long long n) {
  long long i = ((long long)blockIdx.x * blockDim.x + threadIdx.x) * 4;
  if (i + 3 < n) {
    float4 v = *(const float4*)(src + i);
    dst[i + 0] = (__bf16)v.x;
    dst[i + 1] = (__bf16)v.y;
    dst[i + 2] = (__bf16)v.z;
    dst[i + 3] = (__bf16)v.w;
  } else {
    for (long long j = i; j < n; ++j) dst[j] = (__bf16)src[j];
  }
}

// ---------------- transpose + cast: src[K][N] f32 -> dst[N][K] bf16 ----------------
__global__ void fa_transpose_cast_kernel(const float* __restrict__ src, __bf16* __restrict__ dst,
                                         int K, int N) {
  long long i = (long long)blockIdx.x * blockDim.x + threadIdx.x;
  long long total = (long long)K * N;
  if (i < total) {
    int k = (int)(i / N);
    int n = (int)(i % N);
    dst[(size_t)n * K + k] = (__bf16)src[i];
  }
}

// ---------------- generic bf16 WMMA GEMM, async-DMA double-buffered ----------------
// C[m][n] = (sum_{w<taps, k<Ktap} A[m + w - (taps-1)][k] * Bt[n][w*Ktap + k] + bias[n]) * out_scale
// Rows with ((m & tmod_mask) + w - (taps-1)) < 0 read zeros (causal conv left padding)
// via per-lane address redirect into the zeroed scratch `zbuf`.
__global__ __launch_bounds__(256) void fa_gemm_kernel(
    const __bf16* __restrict__ A, int lda,
    const __bf16* __restrict__ Bt,
    const float* __restrict__ bias, float out_scale,
    void* __restrict__ C, int ldc, int c_is_f32,
    int M, int N, int Ktap_log2, int taps, int tmod_mask,
    const __bf16* __restrict__ zbuf) {
  __shared__ __bf16 Asub[2][128][40];
  __shared__ __bf16 Bsub[2][128][40];

  const int tid   = threadIdx.x;
  const int mTile = blockIdx.y * 128;
  const int nTile = blockIdx.x * 128;

  const int wave = tid >> 5, lane = tid & 31;
  const int wm = wave >> 2, wn = wave & 3;      // 2 x 4 wave grid -> 64x32 per wave
  const int lrow = lane & 15, lhalf = lane >> 4;
  const int r = tid >> 1, seg = tid & 1;        // tile staging: 128 rows x 2 halves

  const int Ktap = 1 << Ktap_log2;
  const int Ktot = taps << Ktap_log2;
  const int back = taps - 1;

  // Hoisted, row-invariant staging state.
  const int   tbase = ((mTile + r) & tmod_mask) - back;            // + w >= 0 -> row valid
  const __bf16* aRow = A + (size_t)(mTile + r - back) * lda + seg * 16;
  const __bf16* bRow = Bt + (size_t)(nTile + r) * Ktot + seg * 16;

  v8f acc[4][2];
#pragma unroll
  for (int mi = 0; mi < 4; ++mi)
#pragma unroll
    for (int ni = 0; ni < 2; ++ni) acc[mi][ni] = vzero8f();

  auto issue = [&](int kk, int buf) {
    const int w  = kk >> Ktap_log2;
    const int kb = kk & (Ktap - 1);
    const bool ok = (tbase + w) >= 0;
    const __bf16* ap = ok ? (aRow + ((size_t)w << Ktap_log2) * 0 + (size_t)w * lda + kb) : zbuf;
    unsigned la = (unsigned)(size_t)(&Asub[buf][r][seg * 16]);
    async_copy_b128(la,      ap);
    async_copy_b128(la + 16, ap + 8);
    const __bf16* bp = bRow + kk;
    unsigned lb = (unsigned)(size_t)(&Bsub[buf][r][seg * 16]);
    async_copy_b128(lb,      bp);
    async_copy_b128(lb + 16, bp + 8);
  };

  issue(0, 0);
  const int nk = Ktot >> 5;
  for (int it = 0; it < nk; ++it) {
    const int buf = it & 1;
    wait_async0();        // our tile slice has landed in LDS
    __syncthreads();      // everyone's slice landed; prior reads of buf^1 done
    if (it + 1 < nk) issue((it + 1) << 5, buf ^ 1);  // DMA next tile during compute

    v16bf aF[4], bF[2];
#pragma unroll
    for (int mi = 0; mi < 4; ++mi)
      aF[mi] = ld_v16(&Asub[buf][wm * 64 + mi * 16 + lrow][lhalf * 16]);
#pragma unroll
    for (int ni = 0; ni < 2; ++ni)
      bF[ni] = ld_v16(&Bsub[buf][wn * 32 + ni * 16 + lrow][lhalf * 16]);

#pragma unroll
    for (int mi = 0; mi < 4; ++mi)
#pragma unroll
      for (int ni = 0; ni < 2; ++ni)
        acc[mi][ni] = wmma_bf16(aF[mi], bF[ni], acc[mi][ni]);
  }

  // epilogue: C layout -> lane l, vgpr v holds (M = v + 8*(l/16), N = l%16)
  float* Cf = (float*)C;
  __bf16* Cb = (__bf16*)C;
#pragma unroll
  for (int mi = 0; mi < 4; ++mi) {
#pragma unroll
    for (int ni = 0; ni < 2; ++ni) {
#pragma unroll
      for (int v = 0; v < 8; ++v) {
        int row = mTile + wm * 64 + mi * 16 + v + 8 * lhalf;
        int col = nTile + wn * 32 + ni * 16 + lrow;
        float cv = (acc[mi][ni][v] + bias[col]) * out_scale;
        if (c_is_f32) Cf[(size_t)row * ldc + col] = cv;
        else          Cb[(size_t)row * ldc + col] = (__bf16)cv;
      }
    }
  }
}

// ---------------- fused block-local causal attention ----------------
// One workgroup per (block b of 256 rows, head h). qh is pre-scaled by 1/sqrt(64).
__global__ __launch_bounds__(256) void fa_attn_kernel(
    const __bf16* __restrict__ qh, const __bf16* __restrict__ kh,
    const __bf16* __restrict__ vh, __bf16* __restrict__ ctx) {
  const int b = blockIdx.x;   // 0..63  (batch*16 blocks)
  const int h = blockIdx.y;   // 0..7

  __shared__ __bf16 vT[64][272];        // V transposed: [kdim][s], padded
  __shared__ __bf16 stage[8][16][48];   // per-wave attn A-layout staging

  const int tid = threadIdx.x;

  // stage V transposed (each thread owns one s-row)
  {
    const int s = tid;
    const __bf16* vrow = vh + ((size_t)(b * 256 + s)) * 512 + h * 64;
    const v8bf* v8p = (const v8bf*)vrow;
#pragma unroll
    for (int i = 0; i < 8; ++i) {
      v8bf d = v8p[i];
#pragma unroll
      for (int e = 0; e < 8; ++e) vT[i * 8 + e][s] = d[e];
    }
  }
  __syncthreads();

  const int wave = tid >> 5, lane = tid & 31;
  const int lrow = lane & 15, lhalf = lane >> 4;

  for (int c = 0; c < 2; ++c) {
    const int q0 = wave * 32 + c * 16;   // 16 query rows per chunk, 2 chunks per wave

    // A fragments straight from global (contiguous 16 bf16 per lane)
    v16bf aF[2];
#pragma unroll
    for (int ks = 0; ks < 2; ++ks)
      aF[ks] = ld_v16(qh + ((size_t)(b * 256 + q0 + lrow)) * 512 + h * 64 + ks * 32 + lhalf * 16);

    // scores: 16 tiles of 16x16 over s
    v8f sc[16];
#pragma unroll
    for (int j = 0; j < 16; ++j) sc[j] = vzero8f();
#pragma unroll
    for (int j = 0; j < 16; ++j) {
#pragma unroll
      for (int ks = 0; ks < 2; ++ks) {
        v16bf bF = ld_v16(kh + ((size_t)(b * 256 + j * 16 + lrow)) * 512 + h * 64 + ks * 32 + lhalf * 16);
        sc[j] = wmma_bf16(aF[ks], bF, sc[j]);
      }
    }

    // causal mask + row max
    float mx[8];
#pragma unroll
    for (int v = 0; v < 8; ++v) mx[v] = -3.0e38f;
#pragma unroll
    for (int j = 0; j < 16; ++j) {
#pragma unroll
      for (int v = 0; v < 8; ++v) {
        int row = q0 + v + 8 * lhalf;
        int col = j * 16 + lrow;
        if (col > row) sc[j][v] = -3.0e38f;
        mx[v] = fmaxf(mx[v], sc[j][v]);
      }
    }
#pragma unroll
    for (int v = 0; v < 8; ++v) {
      mx[v] = fmaxf(mx[v], __shfl_xor(mx[v], 1, 32));
      mx[v] = fmaxf(mx[v], __shfl_xor(mx[v], 2, 32));
      mx[v] = fmaxf(mx[v], __shfl_xor(mx[v], 4, 32));
      mx[v] = fmaxf(mx[v], __shfl_xor(mx[v], 8, 32));
    }
    float sum[8];
#pragma unroll
    for (int v = 0; v < 8; ++v) sum[v] = 0.0f;
#pragma unroll
    for (int j = 0; j < 16; ++j) {
#pragma unroll
      for (int v = 0; v < 8; ++v) {
        float e = __expf(sc[j][v] - mx[v]);
        sc[j][v] = e;
        sum[v] += e;
      }
    }
#pragma unroll
    for (int v = 0; v < 8; ++v) {
      sum[v] += __shfl_xor(sum[v], 1, 32);
      sum[v] += __shfl_xor(sum[v], 2, 32);
      sum[v] += __shfl_xor(sum[v], 4, 32);
      sum[v] += __shfl_xor(sum[v], 8, 32);
    }
    float inv[8];
#pragma unroll
    for (int v = 0; v < 8; ++v) inv[v] = 1.0f / sum[v];

    // ctx = attn x V  (K = 256 over s, in 8 steps of 32)
    v8f cacc[4];
#pragma unroll
    for (int ni = 0; ni < 4; ++ni) cacc[ni] = vzero8f();

    for (int jp = 0; jp < 8; ++jp) {
      // stage attn (C layout -> A layout) through LDS as bf16
#pragma unroll
      for (int jj = 0; jj < 2; ++jj) {
        int j = jp * 2 + jj;
#pragma unroll
        for (int v = 0; v < 8; ++v) {
          int m = v + 8 * lhalf;
          stage[wave][m][jj * 16 + lrow] = (__bf16)(sc[j][v] * inv[v]);
        }
      }
      asm volatile("s_wait_dscnt 0" ::: "memory");  // wave-local LDS RAW fence
      v16bf aA = ld_v16(&stage[wave][lrow][lhalf * 16]);
#pragma unroll
      for (int ni = 0; ni < 4; ++ni) {
        v16bf bF = ld_v16(&vT[ni * 16 + lrow][jp * 32 + lhalf * 16]);
        cacc[ni] = wmma_bf16(aA, bF, cacc[ni]);
      }
    }

    // write ctx (bf16) at [row][h*64 + kdim]
#pragma unroll
    for (int ni = 0; ni < 4; ++ni) {
#pragma unroll
      for (int v = 0; v < 8; ++v) {
        int row = b * 256 + q0 + v + 8 * lhalf;
        int col = h * 64 + ni * 16 + lrow;
        ctx[(size_t)row * 512 + col] = (__bf16)cacc[ni][v];
      }
    }
  }
}

extern "C" void kernel_launch(void* const* d_in, const int* in_sizes, int n_in,
                              void* d_out, int out_size, void* d_ws, size_t ws_size,
                              hipStream_t stream) {
  (void)in_sizes; (void)n_in; (void)out_size; (void)ws_size;
  const float* x      = (const float*)d_in[0];
  const float* conv_w = (const float*)d_in[1];
  const float* conv_b = (const float*)d_in[2];
  const float* wq     = (const float*)d_in[3];
  const float* bq     = (const float*)d_in[4];
  const float* wk     = (const float*)d_in[5];
  const float* bk     = (const float*)d_in[6];
  const float* wv     = (const float*)d_in[7];
  const float* bv     = (const float*)d_in[8];
  const float* wo     = (const float*)d_in[9];
  const float* bo     = (const float*)d_in[10];
  const float* proj_w = (const float*)d_in[11];
  const float* proj_b = (const float*)d_in[12];

  const long long MT = 16384;  // N*T = 4*4096 rows
  char* wsp = (char*)d_ws;
  size_t off = 0;
  auto take = [&](size_t elems) -> __bf16* {
    __bf16* p = (__bf16*)(wsp + off);
    off += ((elems * 2 + 255) & ~(size_t)255);
    return p;
  };
  __bf16* zbuf = take(128);                 // 256B zero scratch for conv padding
  __bf16* xb  = take((size_t)MT * 2048);
  __bf16* cwT = take((size_t)1536 * 6144);
  __bf16* wqT = take((size_t)512 * 512);
  __bf16* wkT = take((size_t)512 * 512);
  __bf16* wvT = take((size_t)512 * 512);
  __bf16* woT = take((size_t)512 * 512);
  __bf16* pjT = take((size_t)2048 * 512);
  __bf16* qkv = take((size_t)MT * 1536);
  __bf16* qhb = take((size_t)MT * 512);
  __bf16* khb = take((size_t)MT * 512);
  __bf16* vhb = take((size_t)MT * 512);
  __bf16* ctx = take((size_t)MT * 512);
  __bf16* mha = take((size_t)MT * 512);

  // prep: zero scratch, cast x, transpose-cast weights to [N][K] bf16
  fa_zero_kernel<<<1, 16, 0, stream>>>((float4*)zbuf);
  {
    long long n = MT * 2048;
    int grid = (int)((n / 4 + 255) / 256);
    fa_cast_kernel<<<grid, 256, 0, stream>>>(x, xb, n);
  }
  auto tp = [&](const float* s, __bf16* dptr, int K, int N) {
    long long tot = (long long)K * N;
    int grid = (int)((tot + 255) / 256);
    fa_transpose_cast_kernel<<<grid, 256, 0, stream>>>(s, dptr, K, N);
  };
  tp(conv_w, cwT, 6144, 1536);
  tp(wq, wqT, 512, 512);
  tp(wk, wkT, 512, 512);
  tp(wv, wvT, 512, 512);
  tp(wo, woT, 512, 512);
  tp(proj_w, pjT, 512, 2048);

  // causal conv1d as 3-tap K-fused GEMM: qkv = im2col(x) @ conv_w + conv_b
  fa_gemm_kernel<<<dim3(1536 / 128, 16384 / 128), 256, 0, stream>>>(
      xb, 2048, cwT, conv_b, 1.0f, qkv, 1536, 0, 16384, 1536, 11, 3, 4095, zbuf);

  // Q/K/V projections (scale 1/sqrt(64) folded into Q)
  fa_gemm_kernel<<<dim3(4, 128), 256, 0, stream>>>(
      qkv + 0,    1536, wqT, bq, 0.125f, qhb, 512, 0, 16384, 512, 9, 1, 4095, zbuf);
  fa_gemm_kernel<<<dim3(4, 128), 256, 0, stream>>>(
      qkv + 512,  1536, wkT, bk, 1.0f,   khb, 512, 0, 16384, 512, 9, 1, 4095, zbuf);
  fa_gemm_kernel<<<dim3(4, 128), 256, 0, stream>>>(
      qkv + 1024, 1536, wvT, bv, 1.0f,   vhb, 512, 0, 16384, 512, 9, 1, 4095, zbuf);

  // block-local causal attention
  fa_attn_kernel<<<dim3(64, 8), 256, 0, stream>>>(qhb, khb, vhb, ctx);

  // output projection + final projection (f32 out)
  fa_gemm_kernel<<<dim3(4, 128), 256, 0, stream>>>(
      ctx, 512, woT, bo, 1.0f, mha, 512, 0, 16384, 512, 9, 1, 4095, zbuf);
  fa_gemm_kernel<<<dim3(16, 128), 256, 0, stream>>>(
      mha, 512, pjT, proj_b, 1.0f, d_out, 2048, 1, 16384, 2048, 9, 1, 4095, zbuf);
}
